// LSTMModel_14010183319684
// MI455X (gfx1250) — compile-verified, matrix operand order checked by reading, not run on hardware
//
#include <hip/hip_runtime.h>
#include <hip/hip_bf16.h>

// ---------------- problem constants ----------------
#define BATCH 256
#define TLEN  2048
#define NIN   100
#define NH    100
#define G4    400          // 4*H gate width
#define KP    232          // padded row stride (bf16 elems) for xh / W in LDS (16B aligned, >=224)
#define HOFF  112          // K offset where h-part starts (x part padded 100->112; k=100 is bias col)
#define KT    7            // K tiles of 32 covering 224
#define GS    404          // gates LDS row stride (floats), padded vs 400 to avoid bank conflicts
#define BT    16           // batch rows per workgroup
#define NWAVE 5            // waves per workgroup (5 * 5 tiles = 25 N-tiles = 400 cols)
#define NTPW  5            // N-tiles per wave
#define BLK   (NWAVE * 32) // 160 threads

typedef __attribute__((ext_vector_type(16))) __bf16 v16bf;
typedef __attribute__((ext_vector_type(8)))  float  v8f;

union Frag { uint4 q[2]; v16bf v; };

static __device__ __forceinline__ unsigned short f2bf(float f) {
    unsigned int u = __float_as_uint(f);
    unsigned int r = (u + 0x7FFFu + ((u >> 16) & 1u)) >> 16;   // RNE
    return (unsigned short)r;
}
static __device__ __forceinline__ float sigm(float x) {
    return 1.0f / (1.0f + __expf(-x));
}
static __device__ __forceinline__ float tanh_(float x) {
    x = fminf(15.0f, fmaxf(-15.0f, x));
    float e = __expf(2.0f * x);
    return (e - 1.0f) / (e + 1.0f);
}

__global__ void __launch_bounds__(BLK, 1)
lstm_persist_kernel(const float* __restrict__ x,
                    const float* __restrict__ W_ih,
                    const float* __restrict__ W_hh,
                    const float* __restrict__ b_ih,
                    const float* __restrict__ b_hh,
                    const float* __restrict__ W_lin,
                    const float* __restrict__ b_lin,
                    float* __restrict__ out) {
    extern __shared__ char smem[];
    unsigned short* Wl    = (unsigned short*)smem;        // [G4][KP] bf16: [W_ih | bias | pad | W_hh | pad]
    unsigned short* xh    = Wl + G4 * KP;                 // [BT][KP] bf16: [x_t | 1 | pad | h | pad]
    float*          gates = (float*)(xh + BT * KP);       // [BT][GS] fp32 (reused for final h)
    float*          cbuf  = gates + BT * GS;              // [BT][NH] fp32 cell state

    const int tid  = threadIdx.x;
    const int lane = tid & 31;
    const int wid  = tid >> 5;          // 0..4
    const int l15  = lane & 15;
    const int half = lane >> 4;         // 0/1
    const int b0   = blockIdx.x * BT;   // first batch row of this WG

    // ---- one-time staging: weights+bias -> LDS (bf16, concatenated K), zero state ----
    for (int idx = tid; idx < G4 * KP; idx += BLK) {
        int n = idx / KP, k = idx % KP;
        float v = 0.0f;
        if (k < NIN)                       v = W_ih[n * NIN + k];
        else if (k == NIN)                 v = b_ih[n] + b_hh[n];   // bias folded as K column
        else if (k >= HOFF && k < HOFF+NH) v = W_hh[n * NH + (k - HOFF)];
        Wl[idx] = f2bf(v);
    }
    for (int idx = tid; idx < BT * KP; idx += BLK) xh[idx] = 0;      // x=0, h=0, pads=0
    for (int idx = tid; idx < BT * NH; idx += BLK) cbuf[idx] = 0.0f;
    if (tid < BT) xh[tid * KP + NIN] = 0x3F80;                       // constant-1 bias column
    // per-thread 32-bit element offsets for streaming x (GVS: SGPR base + VGPR offset)
    unsigned xoff[10];
#pragma unroll
    for (int s = 0; s < 10; ++s) {
        int e = tid + s * BLK;
        xoff[s] = (unsigned)((b0 + e / NIN) * (TLEN * NIN) + (e % NIN));
    }
    const unsigned poff = (unsigned)((b0 + l15) * (TLEN * NIN));
    // stage x_0 (exactly 10 elements per thread: 16*100/160)
#pragma unroll
    for (int s = 0; s < 10; ++s) {
        int e = tid + s * BLK;
        xh[(e / NIN) * KP + (e % NIN)] = f2bf(x[xoff[s]]);
    }
    __syncthreads();

    // ---------------- recurrence over T timesteps ----------------
    for (int t = 0; t < TLEN; ++t) {
        // issue next-step x loads early (uniform scalar base + 32-bit voffset)
        float xv[10];
        if (t + 1 < TLEN) {
            const float* xt = x + (size_t)(t + 1) * NIN;
#pragma unroll
            for (int s = 0; s < 10; ++s) xv[s] = xt[xoff[s]];
        }
        if (t + 2 < TLEN)
            __builtin_prefetch(x + (size_t)(t + 2) * NIN + poff, 0, 1);

        // A fragments for this step: [x_t | 1 | h_{t-1}] rows, shared across all N-tiles
        Frag afr[KT];
#pragma unroll
        for (int kt = 0; kt < KT; ++kt) {
            const unsigned short* base = &xh[l15 * KP + kt * 32 + half * 8];
            afr[kt].q[0] = *(const uint4*)(base);
            afr[kt].q[1] = *(const uint4*)(base + 16);
        }
        // 5 concurrent accumulators over a fused k-loop: per k-step, 10 ds_load_b128
        // feed 5 INDEPENDENT WMMAs -> LDS latency and matrix-pipe latency overlap,
        // and no D-register aliasing hazards (no v_nop stalls).
        v8f acc[NTPW];
#pragma unroll
        for (int n = 0; n < NTPW; ++n)
            acc[n] = (v8f){0.f, 0.f, 0.f, 0.f, 0.f, 0.f, 0.f, 0.f};
#pragma unroll
        for (int kt = 0; kt < KT; ++kt) {
            Frag bf[NTPW];
#pragma unroll
            for (int n = 0; n < NTPW; ++n) {
                const unsigned short* wb =
                    &Wl[(wid * (NTPW * 16) + n * 16 + l15) * KP + kt * 32 + half * 8];
                bf[n].q[0] = *(const uint4*)(wb);
                bf[n].q[1] = *(const uint4*)(wb + 16);
            }
#pragma unroll
            for (int n = 0; n < NTPW; ++n)
                acc[n] = __builtin_amdgcn_wmma_f32_16x16x32_bf16(
                            false, afr[kt].v, false, bf[n].v, (short)0, acc[n], false, false);
        }
#pragma unroll
        for (int n = 0; n < NTPW; ++n) {
            const int nrow = wid * (NTPW * 16) + n * 16 + l15;
#pragma unroll
            for (int r = 0; r < 8; ++r)              // C/D layout: M = r + 8*half, N = nrow
                gates[(r + 8 * half) * GS + nrow] = acc[n][r];
        }
        __syncthreads();

        // fused LSTM cell (i,f,g,o PyTorch order), h -> bf16 A-operand for next step
#pragma unroll
        for (int s = 0; s < 10; ++s) {
            int e = tid + s * BLK;
            int row = e / NH, j = e % NH;
            float gi = sigm (gates[row * GS + j]);
            float gf = sigm (gates[row * GS + NH  + j]);
            float gg = tanh_(gates[row * GS + 2*NH + j]);
            float go = sigm (gates[row * GS + 3*NH + j]);
            float c  = gf * cbuf[row * NH + j] + gi * gg;
            cbuf[row * NH + j] = c;
            float h = go * tanh_(c);
            xh[row * KP + HOFF + j] = f2bf(h);
            if (t == TLEN - 1) gates[row * GS + j] = h;       // stash fp32 h_final
            if (t + 1 < TLEN) xh[row * KP + j] = f2bf(xv[s]); // stage x_{t+1}
        }
        __syncthreads();
    }

    // ---------------- final linear (H->2) + softmax ----------------
    if (tid < BT) {
        const int row = tid;
        float d0 = b_lin[0], d1 = b_lin[1];
        for (int j = 0; j < NH; ++j) {
            float h = gates[row * GS + j];
            d0 += h * W_lin[j];
            d1 += h * W_lin[NH + j];
        }
        float mx = fmaxf(d0, d1);
        float e0 = __expf(d0 - mx), e1 = __expf(d1 - mx);
        float inv = 1.0f / (e0 + e1);
        out[(size_t)(b0 + row) * 2 + 0] = e0 * inv;
        out[(size_t)(b0 + row) * 2 + 1] = e1 * inv;
    }
}

extern "C" void kernel_launch(void* const* d_in, const int* in_sizes, int n_in,
                              void* d_out, int out_size, void* d_ws, size_t ws_size,
                              hipStream_t stream) {
    const float* xp    = (const float*)d_in[0];
    const float* W_ih  = (const float*)d_in[1];
    const float* W_hh  = (const float*)d_in[2];
    const float* b_ih  = (const float*)d_in[3];
    const float* b_hh  = (const float*)d_in[4];
    const float* W_lin = (const float*)d_in[5];
    const float* b_lin = (const float*)d_in[6];
    float* out = (float*)d_out;

    const size_t lds_bytes = (size_t)G4 * KP * 2   // weights (+bias col) bf16
                           + (size_t)BT * KP * 2   // xh bf16
                           + (size_t)BT * GS * 4   // gates fp32
                           + (size_t)BT * NH * 4;  // cell fp32  (~225 KB < 320 KB/WGP)

    lstm_persist_kernel<<<dim3(BATCH / BT), dim3(BLK), lds_bytes, stream>>>(
        xp, W_ih, W_hh, b_ih, b_hh, W_lin, b_lin, out);
}
